// SelfAttentionSAG_18451179504153
// MI455X (gfx1250) — compile-verified, matrix operand order checked by reading, not run on hardware
//
#include <hip/hip_runtime.h>
#include <hip/hip_bf16.h>
#include <math.h>

// ---------------------------------------------------------------- constants
#define NN    50000      // nodes
#define EE    800000     // edges
#define DD    128        // channels
#define KK    25000      // kept nodes (ratio 0.5)
#define PADK  25008      // KK padded to multiple of 16 for WMMA tiles
#define OUTC  128
#define BN_EPS 1e-5f

// d_out layout (floats): y | edge_index_out | batch_out | perm | topv | attn
#define Y_OFF     0
#define EDGE_OFF  (KK * OUTC)             // 3,200,000
#define BATCH_OFF (EDGE_OFF + 2 * KK)     // 3,250,000
#define PERM_OFF  (BATCH_OFF + KK)        // 3,275,000
#define TOPV_OFF  (PERM_OFF + KK)         // 3,300,000
#define ATTN_OFF  (TOPV_OFF + KK)         // 3,325,000

// workspace layout (bytes)
#define AGG_OFF    0ULL
#define H_OFF      25600000ULL
#define H2_OFF     51200000ULL
#define XS_OFF     76800000ULL            // PADK*128*4 = 12,804,096
#define YB_OFF     89604096ULL
#define SCORE_OFF  102408192ULL
#define KEYA_OFF   102608192ULL
#define KEYB_OFF   103008192ULL
#define RANK_OFF   103408192ULL
#define STATS_OFF  103608192ULL

// ---------------------------------------------------------------- WMMA types
typedef float  v8f  __attribute__((ext_vector_type(8)));
typedef float  v2f  __attribute__((ext_vector_type(2)));
typedef __bf16 v16bf __attribute__((ext_vector_type(16)));

#if defined(__has_builtin)
#  if __has_builtin(__builtin_amdgcn_wmma_f32_16x16x4_f32)
#    define USE_F32_WMMA 1
#  endif
#endif
#ifndef USE_F32_WMMA
#  define USE_F32_WMMA 0
#endif

// One wave accumulates a 16x16 tile of  As(16x128, LDS) @ W(128x128, global).
// A layout (f32 16x4): lane<16 -> K=k0,k0+1 ; lane>=16 -> K=k0+2,k0+3, M=lane&15.
// B/C/D: N = lane&15; C VGPR j holds row (lane>=16 ? 8 : 0)+j.
__device__ inline void mma_tile128(const float* As, const float* __restrict__ W,
                                   int n0, int lane, v8f& acc) {
  const int m    = lane & 15;   // M index for A, N index for B
  const int half = lane >> 4;
#if USE_F32_WMMA
  const int kb = half * 2;
  #pragma unroll
  for (int k0 = 0; k0 < 128; k0 += 4) {
    v2f a, b;
    a.x = As[m * 128 + k0 + kb];
    a.y = As[m * 128 + k0 + kb + 1];
    b.x = W[(k0 + kb)     * 128 + n0 + m];
    b.y = W[(k0 + kb + 1) * 128 + n0 + m];
    acc = __builtin_amdgcn_wmma_f32_16x16x4_f32(false, a, false, b,
                                                (short)0, acc, false, false);
  }
#else
  const int kb = half * 8;
  #pragma unroll
  for (int k0 = 0; k0 < 128; k0 += 32) {
    v16bf a, b;
    #pragma unroll
    for (int j = 0; j < 8; ++j) {
      const int kk = (j < 4) ? (kb + 2 * j) : (16 + kb + 2 * (j - 4));
      a[2 * j]     = (__bf16)As[m * 128 + k0 + kk];
      a[2 * j + 1] = (__bf16)As[m * 128 + k0 + kk + 1];
      b[2 * j]     = (__bf16)W[(k0 + kk)     * 128 + n0 + m];
      b[2 * j + 1] = (__bf16)W[(k0 + kk + 1) * 128 + n0 + m];
    }
    acc = __builtin_amdgcn_wmma_f32_16x16x32_bf16(false, a, false, b,
                                                  (short)0, acc, false, false);
  }
#endif
}

// Y[r0..r0+15, :128] = A@W (+ A2@W2) + bias.  grid.x = rows/16, block = 256 (8 waves).
__global__ __launch_bounds__(256)
void gemm128_kernel(const float* __restrict__ A,  const float* __restrict__ W,
                    const float* __restrict__ A2, const float* __restrict__ W2,
                    const float* __restrict__ bias, float* __restrict__ Y) {
  __shared__ float As[16 * 128];
  __shared__ float As2[16 * 128];
  const int r0 = blockIdx.x * 16;
  {  // tile is contiguous in row-major memory -> trivial float4 staging
    const float4* Ag = (const float4*)(A + (size_t)r0 * 128);
    float4* AsV = (float4*)As;
    for (int i = threadIdx.x; i < 512; i += 256) AsV[i] = Ag[i];
    if (A2) {
      const float4* Ag2 = (const float4*)(A2 + (size_t)r0 * 128);
      float4* AsV2 = (float4*)As2;
      for (int i = threadIdx.x; i < 512; i += 256) AsV2[i] = Ag2[i];
    }
  }
  __syncthreads();
  const int lane = threadIdx.x & 31;
  const int n0   = (threadIdx.x >> 5) * 16;   // wave -> N tile
  v8f acc;
  #pragma unroll
  for (int j = 0; j < 8; ++j) acc[j] = 0.0f;
  mma_tile128(As, W, n0, lane, acc);
  if (A2) mma_tile128(As2, W2, n0, lane, acc);
  const int n  = lane & 15;
  const int m0 = (lane >> 4) * 8;
  const float bv = bias ? bias[n0 + n] : 0.0f;
  #pragma unroll
  for (int j = 0; j < 8; ++j)
    Y[(size_t)(r0 + m0 + j) * 128 + n0 + n] = acc[j] + bv;
}

// ---------------------------------------------------------------- graph aggregation
__global__ __launch_bounds__(256)
void scatter_add_kernel(const float* __restrict__ x, const int* __restrict__ ei,
                        float* __restrict__ agg) {
  const long long idx = (long long)blockIdx.x * 256 + threadIdx.x;
  if (idx >= (long long)EE * 32) return;
  const int e = (int)(idx >> 5);
  const int c = ((int)idx & 31) * 4;
  const int s = ei[e];
  const int d = ei[EE + e];
  const float4 v = *(const float4*)&x[(size_t)s * 128 + c];
  float* p = &agg[(size_t)d * 128 + c];
  atomicAdd(p + 0, v.x); atomicAdd(p + 1, v.y);
  atomicAdd(p + 2, v.z); atomicAdd(p + 3, v.w);
}

// ---------------------------------------------------------------- score + sort keys
__global__ __launch_bounds__(256)
void score_kernel(const float* __restrict__ h2, const float* __restrict__ w_sel,
                  float* __restrict__ score, unsigned long long* __restrict__ keys) {
  const int row  = blockIdx.x * 8 + (threadIdx.x >> 5);
  if (row >= NN) return;
  const int lane = threadIdx.x & 31;
  const float4 v = *(const float4*)&h2[(size_t)row * 128 + lane * 4];
  float s = v.x + v.y + v.z + v.w;
  for (int o = 16; o; o >>= 1) s += __shfl_xor(s, o, 32);
  if (lane == 0) {
    const float w  = w_sel[0];
    const float sc = tanhf(s * w / fabsf(w));   // == tanh(score_raw/||w||)
    score[row] = sc;
    unsigned u = __float_as_uint(sc);
    u = (u & 0x80000000u) ? ~u : (u | 0x80000000u);   // ascending monotone
    const unsigned dk = ~u;                            // descending score
    keys[row] = ((unsigned long long)dk << 16) | (unsigned)row; // tie: low idx first
  }
}

// single-workgroup stable LSD radix sort, 5-bit digits, 10 passes (48-bit keys).
#define RBITS 5
#define RADIX 32
#define SORTP 10
__global__ __launch_bounds__(256)
void radix_sort_kernel(unsigned long long* keysA, unsigned long long* keysB) {
  __shared__ unsigned int hist[RADIX][256];   // 32 KB
  __shared__ unsigned int digitBase[RADIX];
  const int t = threadIdx.x;
  const int chunk = (NN + 255) / 256;
  const int beg = t * chunk;
  const int end = (beg + chunk < NN) ? beg + chunk : NN;
  unsigned long long* src = keysA;
  unsigned long long* dst = keysB;
  for (int p = 0; p < SORTP; ++p) {
    const int shift = p * RBITS;
    for (int d = 0; d < RADIX; ++d) hist[d][t] = 0;
    __syncthreads();
    for (int i = beg; i < end; ++i)
      hist[(unsigned)((src[i] >> shift) & (RADIX - 1))][t]++;
    __syncthreads();
    if (t < RADIX) {            // digit-major, thread-minor exclusive prefix
      unsigned run = 0;
      for (int j = 0; j < 256; ++j) { unsigned tmp = hist[t][j]; hist[t][j] = run; run += tmp; }
      digitBase[t] = run;
    }
    __syncthreads();
    if (t == 0) {
      unsigned run = 0;
      for (int d = 0; d < RADIX; ++d) { unsigned tmp = digitBase[d]; digitBase[d] = run; run += tmp; }
    }
    __syncthreads();
    for (int i = beg; i < end; ++i) {      // sequential per thread -> stable
      const unsigned long long k = src[i];
      const unsigned d = (unsigned)((k >> shift) & (RADIX - 1));
      dst[digitBase[d] + hist[d][t]++] = k;
    }
    __syncthreads();
    unsigned long long* tmp = src; src = dst; dst = tmp;
  }  // 10 passes (even) -> sorted result in keysA
}

// ---------------------------------------------------------------- selection outputs
__global__ __launch_bounds__(256)
void rank_out_kernel(const unsigned long long* __restrict__ sorted,
                     const float* __restrict__ score, const int* __restrict__ batch,
                     unsigned int* __restrict__ rank, float* __restrict__ out) {
  const int j = blockIdx.x * 256 + threadIdx.x;
  if (j >= NN) return;
  const unsigned node = (unsigned)(sorted[j] & 0xFFFFull);
  rank[node] = (unsigned)j;
  if (j < KK) {
    out[PERM_OFF  + j] = (float)node;
    out[TOPV_OFF  + j] = score[node];
    out[BATCH_OFF + j] = (float)batch[node];
  }
}

__global__ __launch_bounds__(256)
void xs_gather_kernel(const unsigned long long* __restrict__ sorted,
                      const float* __restrict__ score, const float* __restrict__ h2,
                      float* __restrict__ xs) {
  const int idx = blockIdx.x * 256 + threadIdx.x;
  if (idx >= KK * 32) return;
  const int j = idx >> 5;
  const int c = (idx & 31) * 4;
  const unsigned node = (unsigned)(sorted[j] & 0xFFFFull);
  const float tv = score[node];
  float4 v = *(const float4*)&h2[(size_t)node * 128 + c];
  v.x *= tv; v.y *= tv; v.z *= tv; v.w *= tv;
  *(float4*)&xs[(size_t)j * 128 + c] = v;
}

// kept = ascending node ids with rank<K; edge rows are their ranks (== cluster ids)
__global__ __launch_bounds__(256)
void compact_edges_kernel(const unsigned int* __restrict__ rank,
                          float* __restrict__ out_edge) {
  __shared__ unsigned int tsum[256];
  const int t = threadIdx.x;
  const int chunk = (NN + 255) / 256;
  const int beg = t * chunk;
  const int end = (beg + chunk < NN) ? beg + chunk : NN;
  unsigned cnt = 0;
  for (int i = beg; i < end; ++i) cnt += (rank[i] < (unsigned)KK);
  tsum[t] = cnt;
  __syncthreads();
  if (t == 0) {
    unsigned run = 0;
    for (int j = 0; j < 256; ++j) { unsigned tmp = tsum[j]; tsum[j] = run; run += tmp; }
  }
  __syncthreads();
  unsigned pos = tsum[t];
  for (int i = beg; i < end; ++i) {
    const unsigned r = rank[i];
    if (r < (unsigned)KK) {
      out_edge[pos]      = (float)r;
      out_edge[KK + pos] = (float)r;
      ++pos;
    }
  }
}

// ---------------------------------------------------------------- batchnorm + misc
__global__ __launch_bounds__(256)
void bn_stats_kernel(const float* __restrict__ y, float* __restrict__ stats) {
  const int c = blockIdx.x;                       // one column per block
  __shared__ float s1[256], s2[256];
  float a = 0.f, b = 0.f;
  for (int r = threadIdx.x; r < KK; r += 256) {
    const float v = y[(size_t)r * 128 + c];
    a += v; b += v * v;
  }
  s1[threadIdx.x] = a; s2[threadIdx.x] = b;
  __syncthreads();
  for (int o = 128; o; o >>= 1) {
    if ((int)threadIdx.x < o) { s1[threadIdx.x] += s1[threadIdx.x + o]; s2[threadIdx.x] += s2[threadIdx.x + o]; }
    __syncthreads();
  }
  if (threadIdx.x == 0) {
    const float mu  = s1[0] / (float)KK;
    float var = s2[0] / (float)KK - mu * mu;      // biased variance
    if (var < 0.f) var = 0.f;
    stats[c] = mu; stats[128 + c] = var;
  }
}

__global__ __launch_bounds__(256)
void bn_apply_kernel(const float* __restrict__ y, const float* __restrict__ stats,
                     const float* __restrict__ gamma, const float* __restrict__ beta,
                     float* __restrict__ out) {
  const int i = blockIdx.x * 256 + threadIdx.x;
  if (i >= KK * OUTC) return;
  const int c = i & 127;
  const float v = (y[i] - stats[c]) * rsqrtf(stats[128 + c] + BN_EPS) * gamma[c] + beta[c];
  out[Y_OFF + i] = v > 0.f ? v : 0.f;
}

__global__ __launch_bounds__(256)
void ones_kernel(float* __restrict__ out) {       // attn == softmax over singleton
  const int i = blockIdx.x * 256 + threadIdx.x;
  if (i < NN) out[i] = 1.0f;
}

// ---------------------------------------------------------------- launch
extern "C" void kernel_launch(void* const* d_in, const int* in_sizes, int n_in,
                              void* d_out, int out_size, void* d_ws, size_t ws_size,
                              hipStream_t stream) {
  (void)in_sizes; (void)n_in; (void)out_size; (void)ws_size;
  const float* x      = (const float*)d_in[0];
  const int*   ei     = (const int*)  d_in[1];
  const int*   batch  = (const int*)  d_in[2];
  const float* W_rel  = (const float*)d_in[3];
  const float* b_rel  = (const float*)d_in[4];
  const float* W_root = (const float*)d_in[5];
  // d_in[6..9] (Wq,bq,Wk,bk) are dead: softmax over a singleton axis is identically 1.
  const float* Wv     = (const float*)d_in[10];
  const float* bv     = (const float*)d_in[11];
  const float* w_sel  = (const float*)d_in[12];
  const float* Wf     = (const float*)d_in[13];
  const float* bfv    = (const float*)d_in[14];
  const float* gamma  = (const float*)d_in[15];
  const float* beta   = (const float*)d_in[16];
  float* out = (float*)d_out;

  char* ws = (char*)d_ws;
  float*              agg   = (float*)(ws + AGG_OFF);
  float*              h     = (float*)(ws + H_OFF);
  float*              h2    = (float*)(ws + H2_OFF);
  float*              xs    = (float*)(ws + XS_OFF);
  float*              ybuf  = (float*)(ws + YB_OFF);
  float*              score = (float*)(ws + SCORE_OFF);
  unsigned long long* keysA = (unsigned long long*)(ws + KEYA_OFF);
  unsigned long long* keysB = (unsigned long long*)(ws + KEYB_OFF);
  unsigned int*       rank  = (unsigned int*)(ws + RANK_OFF);
  float*              stats = (float*)(ws + STATS_OFF);

  hipMemsetAsync(agg, 0, (size_t)NN * DD * sizeof(float), stream);
  hipMemsetAsync(xs,  0, (size_t)PADK * DD * sizeof(float), stream); // zero pad rows

  const long long sc_total = (long long)EE * 32;
  scatter_add_kernel<<<(unsigned)((sc_total + 255) / 256), 256, 0, stream>>>(x, ei, agg);

  // h = agg@W_rel + x@W_root + b_rel    (two GEMMs fused into one pass)
  gemm128_kernel<<<NN / 16, 256, 0, stream>>>(agg, W_rel, x, W_root, b_rel, h);
  // h2 = h@Wv + bv  (== attention output, attn == 1)
  gemm128_kernel<<<NN / 16, 256, 0, stream>>>(h, Wv, nullptr, nullptr, bv, h2);

  score_kernel<<<(NN + 7) / 8, 256, 0, stream>>>(h2, w_sel, score, keysA);
  radix_sort_kernel<<<1, 256, 0, stream>>>(keysA, keysB);
  rank_out_kernel<<<(NN + 255) / 256, 256, 0, stream>>>(keysA, score, batch, rank, out);
  xs_gather_kernel<<<(KK * 32 + 255) / 256, 256, 0, stream>>>(keysA, score, h2, xs);
  compact_edges_kernel<<<1, 256, 0, stream>>>(rank, out + EDGE_OFF);

  // y = xs@Wf + bf  (rows padded to 25008; pad rows are zero and ignored)
  gemm128_kernel<<<PADK / 16, 256, 0, stream>>>(xs, Wf, nullptr, nullptr, bfv, ybuf);
  bn_stats_kernel<<<OUTC, 256, 0, stream>>>(ybuf, stats);
  bn_apply_kernel<<<(KK * OUTC + 255) / 256, 256, 0, stream>>>(ybuf, stats, gamma, beta, out);
  ones_kernel<<<(NN + 255) / 256, 256, 0, stream>>>(out + ATTN_OFF);
}
